// Lstm2D_58875411693707
// MI455X (gfx1250) — compile-verified, matrix operand order checked by reading, not run on hardware
//
#include <hip/hip_runtime.h>
#include <hip/hip_bf16.h>

// ---------------------------------------------------------------------------
// 2D-LSTM (B=64, NC=512, 32x32 grid of 4x4 patches) for gfx1250 / MI455X.
// Persistent-kernel design: 32 workgroups, weights resident in LDS (bf16),
// v_wmma_f32_16x16x32_bf16 for the per-step GEMM, atomic grid barrier between
// the 1024 sequential steps.
// ---------------------------------------------------------------------------

static constexpr int Bb   = 64;     // batch
static constexpr int NCc  = 512;    // cells
static constexpr int SYy  = 32, SXx = 32;
static constexpr int Tt   = SYy * SXx;          // 1024 steps
static constexpr int INp  = 48;                 // C*P*P
static constexpr int KXx  = 64;                 // x K padded to 2*32
static constexpr int WCOLS = KXx + NCc;         // 576 padded W_ih columns
static constexpr int NBLK = 32;                 // persistent workgroups
static constexpr int NTHR = 256;                // 8 wave32
static constexpr int NSL  = 16;                 // NC slice per workgroup

typedef __attribute__((ext_vector_type(16))) __bf16 v16bf;
typedef __attribute__((ext_vector_type(8)))  float  v8f;

// ---- workspace layout (bytes) ----
static constexpr size_t SZ_WIH  = (size_t)4 * NCc * WCOLS * 2;   // 2.36 MB bf16
static constexpr size_t SZ_WHH  = (size_t)4 * NCc * NCc * 2;     // 2.10 MB bf16
static constexpr size_t SZ_BIAS = (size_t)4 * NCc * 4;           // fused bias f32
static constexpr size_t SZ_XS   = (size_t)Tt * Bb * KXx * 2;     // 8.39 MB bf16
static constexpr size_t SZ_H    = (size_t)Bb * NCc * 2;          // h ping/pong bf16
static constexpr size_t SZ_C    = (size_t)Bb * NCc * 4;          // c f32
static constexpr size_t SZ_ROW  = (size_t)SXx * Bb * NCc * 2;    // row-above ring bf16

static constexpr size_t OFF_WIH  = 0;
static constexpr size_t OFF_WHH  = OFF_WIH + SZ_WIH;
static constexpr size_t OFF_BIAS = OFF_WHH + SZ_WHH;
static constexpr size_t OFF_XS   = OFF_BIAS + SZ_BIAS;
static constexpr size_t OFF_H0   = OFF_XS + SZ_XS;
static constexpr size_t OFF_H1   = OFF_H0 + SZ_H;
static constexpr size_t OFF_C    = OFF_H1 + SZ_H;
static constexpr size_t OFF_ROW  = OFF_C + SZ_C;
static constexpr size_t OFF_BAR  = OFF_ROW + SZ_ROW;
static constexpr size_t STATE_BYTES = (OFF_BAR + 256) - OFF_H0;  // zeroed region

// ---- LDS layout (dynamic) ----
static constexpr size_t LDS_WIH = (size_t)64 * WCOLS * 2;   // 73728
static constexpr size_t LDS_WHH = (size_t)64 * NCc * 2;     // 65536
static constexpr size_t LDS_G   = (size_t)64 * 64 * 4;      // 16384
static constexpr size_t LDS_TOTAL = LDS_WIH + LDS_WHH + LDS_G;  // 155648 < 320K/WGP

// ---------------------------------------------------------------------------
// Prep kernels
// ---------------------------------------------------------------------------
__global__ void prep_weights(const float* __restrict__ Wih,
                             const float* __restrict__ Whh,
                             const float* __restrict__ bih,
                             const float* __restrict__ bhh,
                             __bf16* __restrict__ wih_pad,
                             __bf16* __restrict__ whh_b,
                             float* __restrict__ bias) {
  int idx = blockIdx.x * blockDim.x + threadIdx.x;
  const int n1 = 4 * NCc * WCOLS;
  const int n2 = 4 * NCc * NCc;
  if (idx < n1) {
    int r = idx / WCOLS, c = idx % WCOLS;
    float v = 0.f;
    if (c < INp)       v = Wih[r * (INp + NCc) + c];            // x part
    else if (c >= KXx) v = Wih[r * (INp + NCc) + (c - KXx + INp)]; // prev2 part
    wih_pad[idx] = (__bf16)v;
  } else if (idx < n1 + n2) {
    int k = idx - n1;
    whh_b[k] = (__bf16)Whh[k];
  } else if (idx < n1 + n2 + 4 * NCc) {
    int k = idx - n1 - n2;
    bias[k] = bih[k] + bhh[k];
  }
}

__global__ void prep_xs(const float* __restrict__ batch, __bf16* __restrict__ xs) {
  int idx = blockIdx.x * blockDim.x + threadIdx.x;
  if (idx >= Tt * Bb * KXx) return;
  int k = idx & 63;
  int b = (idx >> 6) & 63;
  int t = idx >> 12;
  int i = t >> 5, j = t & 31;
  float v = 0.f;
  if (k < INp) {
    int c  = k >> 4;          // channel
    int py = (k >> 2) & 3;
    int px = k & 3;
    v = batch[(((size_t)b * 3 + c) * 128 + (i * 4 + py)) * 128 + (j * 4 + px)];
  }
  xs[idx] = (__bf16)v;
}

__global__ void prep_state(unsigned* __restrict__ p, int n) {
  int idx = blockIdx.x * blockDim.x + threadIdx.x;
  if (idx < n) p[idx] = 0u;
}

// ---------------------------------------------------------------------------
// Fragment loaders per CDNA5 ISA 7.12.2 (16-bit A 16x32 / B 32x16):
// half-wave selects K+0 vs K+8; each lane holds K chunks {k0..k0+7, k0+16..+23}.
// ---------------------------------------------------------------------------
union FragU { uint4 u[2]; v16bf v; };

__device__ __forceinline__ v16bf load_frag(const __bf16* base, int stride,
                                           int row0, int k0, int lane) {
  int hsel = (lane & 16);   // +16 bytes (= +8 bf16) for high half-wave
  const char* p = (const char*)(base + (size_t)(row0 + (lane & 15)) * stride + k0);
  FragU f;
  f.u[0] = *(const uint4*)(p + hsel);
  f.u[1] = *(const uint4*)(p + 32 + hsel);
  return f.v;
}

__device__ __forceinline__ float fsig(float x) { return 1.f / (1.f + __expf(-x)); }
__device__ __forceinline__ float ftanh(float x) { return 2.f * fsig(2.f * x) - 1.f; }

#define WMMA_BF16(A, Bm, C) \
  __builtin_amdgcn_wmma_f32_16x16x32_bf16(false, (A), false, (Bm), (short)0, (C), false, false)

// ---------------------------------------------------------------------------
// Persistent 2D-LSTM kernel: blockIdx.x g owns cells [g*16, g*16+16).
// ---------------------------------------------------------------------------
__global__ __launch_bounds__(NTHR)
void lstm2d_wmma(const __bf16* __restrict__ wih, const __bf16* __restrict__ whh,
                 const float* __restrict__ bias, const __bf16* __restrict__ xs,
                 __bf16* __restrict__ h0, __bf16* __restrict__ h1,
                 float* __restrict__ cbuf, __bf16* __restrict__ rowbuf,
                 unsigned* __restrict__ bar, float* __restrict__ out) {
  extern __shared__ char smem[];
  __bf16* lwih = (__bf16*)smem;                       // 64 x 576
  __bf16* lwhh = (__bf16*)(smem + LDS_WIH);           // 64 x 512
  float*  lg   = (float*)(smem + LDS_WIH + LDS_WHH);  // 64(batch) x 64(4 gates x 16)

  const int g   = blockIdx.x;
  const int n0  = g * NSL;
  const int tid = threadIdx.x;

  // ---- stage this WG's weight rows into LDS (kept resident for all steps) ----
  // local row r maps to global row (r>>4)*NC + n0 + (r&15)  (gate-major tiles)
  for (int p = tid; p < 64 * (WCOLS / 2); p += NTHR) {
    int r = p / (WCOLS / 2), c2 = p % (WCOLS / 2);
    int gr = (r >> 4) * NCc + n0 + (r & 15);
    ((unsigned*)lwih)[r * (WCOLS / 2) + c2] =
        ((const unsigned*)wih)[(size_t)gr * (WCOLS / 2) + c2];
  }
  for (int p = tid; p < 64 * (NCc / 2); p += NTHR) {
    int r = p / (NCc / 2), c2 = p % (NCc / 2);
    int gr = (r >> 4) * NCc + n0 + (r & 15);
    ((unsigned*)lwhh)[r * (NCc / 2) + c2] =
        ((const unsigned*)whh)[(size_t)gr * (NCc / 2) + c2];
  }
  __syncthreads();

  const int w    = tid >> 5;
  const int lane = tid & 31;
  const int m    = (w & 3) * 16;           // batch tile row
  const int q0   = (w >> 2) * 2;           // first gate of this wave's pair
  const int q1   = q0 + 1;

  for (int t = 0; t < Tt; ++t) {
    const __bf16* hin  = (t & 1) ? h1 : h0;
    __bf16*       hout = (t & 1) ? h0 : h1;
    const int j = t & 31;

    // publish h(t-1) into the row-above ring buffer (safe: nothing reads
    // column (t-1)&31 again until step t+31, with grid barriers in between)
    if (t > 0) {
      __bf16* rb = rowbuf + (size_t)((t - 1) & 31) * (Bb * NCc);
      #pragma unroll
      for (int u = 0; u < 4; ++u) {
        int e = tid * 4 + u; int b = e >> 4; int nn = e & 15;
        rb[b * NCc + n0 + nn] = hin[b * NCc + n0 + nn];
      }
    }

    const __bf16* xst = xs + (size_t)t * (Bb * KXx);
    const __bf16* pr  = rowbuf + (size_t)j * (Bb * NCc);
    if (t + 1 < Tt)  // pull next step's x tile toward L2/L0
      __builtin_prefetch((const char*)(xs + (size_t)(t + 1) * (Bb * KXx)) + lane * 256, 0, 1);

    v8f acc0 = {}; v8f acc1 = {};

    // --- segment X: K = 0..63 (zero-padded x patch) ---
    #pragma unroll
    for (int kk = 0; kk < KXx; kk += 32) {
      v16bf a  = load_frag(xst, KXx, m, kk, lane);
      v16bf b0 = load_frag(lwih, WCOLS, q0 * 16, kk, lane);
      v16bf b1 = load_frag(lwih, WCOLS, q1 * 16, kk, lane);
      acc0 = WMMA_BF16(a, b0, acc0);
      acc1 = WMMA_BF16(a, b1, acc1);
    }
    // --- segment P: row-above h through W_ih cols 64..575 ---
    #pragma unroll 4
    for (int kk = 0; kk < NCc; kk += 32) {
      v16bf a  = load_frag(pr, NCc, m, kk, lane);
      v16bf b0 = load_frag(lwih, WCOLS, q0 * 16, KXx + kk, lane);
      v16bf b1 = load_frag(lwih, WCOLS, q1 * 16, KXx + kk, lane);
      acc0 = WMMA_BF16(a, b0, acc0);
      acc1 = WMMA_BF16(a, b1, acc1);
    }
    // --- segment H: previous-step h through W_hh ---
    #pragma unroll 4
    for (int kk = 0; kk < NCc; kk += 32) {
      v16bf a  = load_frag(hin, NCc, m, kk, lane);
      v16bf b0 = load_frag(lwhh, NCc, q0 * 16, kk, lane);
      v16bf b1 = load_frag(lwhh, NCc, q1 * 16, kk, lane);
      acc0 = WMMA_BF16(a, b0, acc0);
      acc1 = WMMA_BF16(a, b1, acc1);
    }

    // scatter C/D tiles to LDS (16x16 f32 layout: VGPR r -> row r (+8 for hi half))
    {
      int col = lane & 15;
      int ro  = (lane >> 4) * 8;
      #pragma unroll
      for (int r = 0; r < 8; ++r) {
        lg[(m + ro + r) * 64 + q0 * 16 + col] = acc0[r];
        lg[(m + ro + r) * 64 + q1 * 16 + col] = acc1[r];
      }
    }
    __syncthreads();

    // elementwise LSTM cell update: 4 (b, cell) pairs per thread
    #pragma unroll
    for (int u = 0; u < 4; ++u) {
      int e = tid * 4 + u; int b = e >> 4; int nn = e & 15;
      int gx = b * 64 + nn;
      float iv = lg[gx +  0] + bias[0 * NCc + n0 + nn];
      float fv = lg[gx + 16] + bias[1 * NCc + n0 + nn];
      float gv = lg[gx + 32] + bias[2 * NCc + n0 + nn];
      float ov = lg[gx + 48] + bias[3 * NCc + n0 + nn];
      int ci = b * NCc + n0 + nn;
      float cn = fsig(fv) * cbuf[ci] + fsig(iv) * ftanh(gv);
      float hn = fsig(ov) * ftanh(cn);
      cbuf[ci] = cn;
      hout[ci] = (__bf16)hn;
      // d_out flat layout is (B, T, NC): the reference reshape is a pure view
      out[(size_t)b * (Tt * NCc) + (size_t)t * NCc + n0 + nn] = hn;
    }

    // ---- grid-wide barrier between sequential steps ----
    __syncthreads();
    if (tid == 0) {
      __hip_atomic_fetch_add(bar, 1u, __ATOMIC_RELEASE, __HIP_MEMORY_SCOPE_AGENT);
      unsigned tgt = (unsigned)(t + 1) * NBLK;
      while (__hip_atomic_load(bar, __ATOMIC_ACQUIRE, __HIP_MEMORY_SCOPE_AGENT) < tgt)
        __builtin_amdgcn_s_sleep(2);
    }
    __syncthreads();
  }
}

// ---------------------------------------------------------------------------
extern "C" void kernel_launch(void* const* d_in, const int* in_sizes, int n_in,
                              void* d_out, int out_size, void* d_ws, size_t ws_size,
                              hipStream_t stream) {
  (void)in_sizes; (void)n_in; (void)out_size; (void)ws_size;
  const float* batch = (const float*)d_in[0];
  const float* Wih   = (const float*)d_in[1];
  const float* Whh   = (const float*)d_in[2];
  const float* bih   = (const float*)d_in[3];
  const float* bhh   = (const float*)d_in[4];

  char* ws = (char*)d_ws;
  __bf16*  wih_pad = (__bf16*)(ws + OFF_WIH);
  __bf16*  whh_b   = (__bf16*)(ws + OFF_WHH);
  float*   bias    = (float*)(ws + OFF_BIAS);
  __bf16*  xs      = (__bf16*)(ws + OFF_XS);
  __bf16*  h0      = (__bf16*)(ws + OFF_H0);
  __bf16*  h1      = (__bf16*)(ws + OFF_H1);
  float*   cbuf    = (float*)(ws + OFF_C);
  __bf16*  rowbuf  = (__bf16*)(ws + OFF_ROW);
  unsigned* bar    = (unsigned*)(ws + OFF_BAR);

  // weights / bias -> bf16 (padded), fused bias
  int nw = 4 * NCc * WCOLS + 4 * NCc * NCc + 4 * NCc;
  prep_weights<<<(nw + 255) / 256, 256, 0, stream>>>(Wih, Whh, bih, bhh,
                                                     wih_pad, whh_b, bias);
  // patch extraction -> bf16, zero-padded K
  int nx = Tt * Bb * KXx;
  prep_xs<<<(nx + 255) / 256, 256, 0, stream>>>(batch, xs);
  // zero h/c/rowbuf/barrier
  int nz = (int)(STATE_BYTES / 4);
  prep_state<<<(nz + 255) / 256, 256, 0, stream>>>((unsigned*)(ws + OFF_H0), nz);

  (void)hipFuncSetAttribute((const void*)lstm2d_wmma,
                            hipFuncAttributeMaxDynamicSharedMemorySize,
                            (int)LDS_TOTAL);
  lstm2d_wmma<<<NBLK, NTHR, LDS_TOTAL, stream>>>(wih_pad, whh_b, bias, xs,
                                                 h0, h1, cbuf, rowbuf, bar,
                                                 (float*)d_out);
}
